// VectorNet_80530636800678
// MI455X (gfx1250) — matrix operand order repaired and checked by reading, compiled, not run on hardware
//
#include <hip/hip_runtime.h>
#include <hip/hip_bf16.h>
#include <math.h>
#include <stdint.h>

// ---------------------------------------------------------------------------
// CDNA5 (gfx1250) WMMA helpers.  wave32; D = A(16x32 f16) * B(32x16 f16) + C(f32)
// ---------------------------------------------------------------------------
typedef __attribute__((ext_vector_type(16))) _Float16 v16h;
typedef __attribute__((ext_vector_type(8)))  float    v8f;
typedef __attribute__((ext_vector_type(4)))  unsigned u32x4;
typedef __attribute__((ext_vector_type(8)))  unsigned u32x8;

#define LN_EPS 1e-5f

__device__ __forceinline__ v8f wmma_f16f32(v16h a, v16h b, v8f c) {
  // (neg_a, A, neg_b, B, c_mod, C, reuse_a, reuse_b)
  return __builtin_amdgcn_wmma_f32_16x16x32_f16(false, a, false, b, (short)0, c,
                                                false, false);
}

// ---------------------------------------------------------------------------
// CDNA5 data movers (inline asm: portable across toolchain builtin arities)
// ---------------------------------------------------------------------------
// Async global->LDS copy, 16B per lane, tracked by ASYNCcnt.
__device__ __forceinline__ void async_ld_b128(unsigned lds_off, const void* g) {
  asm volatile("global_load_async_to_lds_b128 %0, %1, off"
               :: "v"(lds_off), "v"((unsigned long)g) : "memory");
}
__device__ __forceinline__ void wait_asynccnt0() {
  asm volatile("s_wait_asynccnt 0x0" ::: "memory");
}

// Tensor Data Mover: 2-D tile (nrows x ncols, 4-byte elements, row stride in
// elements) from global to LDS.  D# groups per CDNA5 ISA.
__device__ __forceinline__ void tdm_load_2d(const void* gptr, void* lptr,
                                            unsigned ncols, unsigned nrows,
                                            unsigned row_stride) {
  unsigned long ga = (unsigned long)gptr;
  unsigned lds = (unsigned)(uintptr_t)lptr;
  u32x4 g0;
  g0[0] = 1u;                                   // count=1 user descriptor
  g0[1] = lds;                                  // lds_addr [63:32]
  g0[2] = (unsigned)ga;                         // global_addr[31:0]
  g0[3] = ((unsigned)(ga >> 32) & 0x01FFFFFFu)  // global_addr[56:32]
          | 0x80000000u;                        // type=2 ("image") [127:126]
  u32x8 g1;
  g1[0] = 0x00020000u;                          // wg_mask=0, data_size=2 (4B)
  g1[1] = (ncols & 0xFFFFu) << 16;              // tensor_dim0[15:0] @ [63:48]
  g1[2] = (ncols >> 16) | ((nrows & 0xFFFFu) << 16);   // dim0 hi | dim1 lo
  g1[3] = (nrows >> 16) | ((ncols & 0xFFFFu) << 16);   // dim1 hi | tile_dim0
  g1[4] = (nrows & 0xFFFFu);                    // tile_dim1 (tile_dim2=0)
  g1[5] = row_stride;                           // tensor_dim0_stride[31:0]
  g1[6] = 0u;                                   // stride hi | dim1_stride lo
  g1[7] = 0u;
  asm volatile("tensor_load_to_lds %0, %1" :: "s"(g0), "s"(g1) : "memory");
}

// ---------------------------------------------------------------------------
// Fragment loaders (ISA §7.12.2 layouts).
// A 16x32 (MxK): lanes 0-15 rows 0..15 with K {0..7,16..23}; lanes 16-31 same
// rows, K {8..15,24..31}.  B 32x16 (KxN): lanes 0-15 K {0..7,16..23}, lanes 16-31
// K {8..15,24..31}, column = n0 + (lane&15).
// ---------------------------------------------------------------------------
__device__ __forceinline__ v16h frag_a_f16(const _Float16* base, int ld, int k0,
                                           int lane) {
  const int half = (lane >> 4) & 1, r = lane & 15;
  v16h a;
#pragma unroll
  for (int e = 0; e < 16; ++e) {
    int k = k0 + ((e >> 3) << 4) + (half << 3) + (e & 7);
    a[e] = base[r * ld + k];
  }
  return a;
}

// f32 LDS source, converted at fragment-build time; k>=kvalid yields 0 (K pad)
__device__ __forceinline__ v16h frag_a_f32(const float* base, int ld, int k0,
                                           int lane, int kvalid) {
  const int half = (lane >> 4) & 1, r = lane & 15;
  v16h a;
#pragma unroll
  for (int e = 0; e < 16; ++e) {
    int k = k0 + ((e >> 3) << 4) + (half << 3) + (e & 7);
    a[e] = (k < kvalid) ? (_Float16)base[r * ld + k] : (_Float16)0.0f;
  }
  return a;
}

__device__ __forceinline__ v16h frag_b_f32(const float* base, int ld, int k0,
                                           int n0, int lane, int kvalid,
                                           int nvalid) {
  const int half = (lane >> 4) & 1, r = lane & 15;
  const int n = n0 + r;
  v16h b;
#pragma unroll
  for (int e = 0; e < 16; ++e) {
    int k = k0 + ((e >> 3) << 4) + (half << 3) + (e & 7);
    b[e] = (k < kvalid && n < nvalid) ? (_Float16)base[k * ld + n]
                                      : (_Float16)0.0f;
  }
  return b;
}

// ---------------------------------------------------------------------------
// Fused  relu(LN(X @ W1 + b1)) @ W2 + b2   for a 16-row tile, one wave/tile.
// A tile (contiguous 16*CIN floats) arrives via global_load_async_to_lds_b128;
// W1/W2 arrive via the Tensor Data Mover.  LayerNorm runs in the WMMA
// accumulators with a 16-lane shfl_xor reduction; sH is the f16 LDS round-trip
// that transposes C-layout -> A-layout for the second GEMM.
// Graph layers: COUT=CIN, OSTRIDE=2*CIN (h-part of concat).  Head: 64->60.
// ---------------------------------------------------------------------------
template <int CIN, int COUT, int OSTRIDE>
__global__ void __launch_bounds__(32)
mlp16_wmma_kernel(const float* __restrict__ xin,
                  const float* __restrict__ W1, const float* __restrict__ b1,
                  const float* __restrict__ g,  const float* __restrict__ be,
                  const float* __restrict__ W2, const float* __restrict__ b2,
                  float* __restrict__ out) {
  constexpr int HID = 64;
  constexpr int KT1 = (CIN + 31) / 32;
  constexpr int NT1 = HID / 16;
  constexpr int KT2 = HID / 32;
  constexpr int NT2 = (COUT + 15) / 16;

  __shared__ float    sAf[16 * CIN];
  __shared__ float    sW1f[CIN * HID];
  __shared__ float    sW2f[HID * COUT];
  __shared__ _Float16 sH[16 * HID];

  const int lane = threadIdx.x & 31;
  const int half = lane >> 4;
  const int r    = lane & 15;
  const int row0 = blockIdx.x * 16;

  // Weights via TDM (wave-scalar, EXEC-independent)
  tdm_load_2d(W1, sW1f, HID, CIN, HID);
  tdm_load_2d(W2, sW2f, COUT, HID, COUT);

  // Activation tile via async copy: one contiguous block of 4*CIN 16B chunks
  {
    const char* gsrc = (const char*)(xin + (long)row0 * CIN);
    const unsigned ldst = (unsigned)(uintptr_t)sAf;
    for (int c = lane; c < 4 * CIN; c += 32)
      async_ld_b128(ldst + c * 16, gsrc + c * 16);
  }
  wait_asynccnt0();
  __builtin_amdgcn_s_wait_tensorcnt(0);
  __syncthreads();

  v8f z = {};
  v8f acc[NT1];
#pragma unroll
  for (int t = 0; t < NT1; ++t) acc[t] = z;
#pragma unroll
  for (int kt = 0; kt < KT1; ++kt) {
    v16h a = frag_a_f32(sAf, CIN, kt * 32, lane, CIN);
#pragma unroll
    for (int t = 0; t < NT1; ++t)
      acc[t] = wmma_f16f32(a, frag_b_f32(sW1f, HID, kt * 32, t * 16, lane,
                                         CIN, HID), acc[t]);
  }

  // + b1, row-wise LayerNorm over 64, affine, ReLU — all in accumulators
#pragma unroll
  for (int t = 0; t < NT1; ++t) {
    float bb = b1[t * 16 + r];
#pragma unroll
    for (int i = 0; i < 8; ++i) acc[t][i] += bb;
  }
#pragma unroll
  for (int i = 0; i < 8; ++i) {
    float s = 0.f, s2 = 0.f;
#pragma unroll
    for (int t = 0; t < NT1; ++t) { float v = acc[t][i]; s += v; s2 += v * v; }
#pragma unroll
    for (int m = 1; m < 16; m <<= 1) {          // reduce within 16-lane group
      s  += __shfl_xor(s,  m, 32);
      s2 += __shfl_xor(s2, m, 32);
    }
    float mu  = s * (1.0f / HID);
    float var = s2 * (1.0f / HID) - mu * mu;
    float rs  = rsqrtf(var + LN_EPS);
    int row = half * 8 + i;
#pragma unroll
    for (int t = 0; t < NT1; ++t) {
      int col = t * 16 + r;
      float v = (acc[t][i] - mu) * rs * g[col] + be[col];
      sH[row * HID + col] = (_Float16)fmaxf(v, 0.0f);
    }
  }
  __syncthreads();   // C-layout -> A-layout transpose through LDS

  v8f acc2[NT2];
#pragma unroll
  for (int t = 0; t < NT2; ++t) acc2[t] = z;
#pragma unroll
  for (int kt = 0; kt < KT2; ++kt) {
    v16h a = frag_a_f16(sH, HID, kt * 32, lane);
#pragma unroll
    for (int t = 0; t < NT2; ++t)
      acc2[t] = wmma_f16f32(a, frag_b_f32(sW2f, COUT, kt * 32, t * 16, lane,
                                          HID, COUT), acc2[t]);
  }
#pragma unroll
  for (int t = 0; t < NT2; ++t) {
    int col = t * 16 + r;
    if (col < COUT) {
      float bb = b2[col];
#pragma unroll
      for (int i = 0; i < 8; ++i)
        out[(long)(row0 + half * 8 + i) * OSTRIDE + col] = acc2[t][i] + bb;
    }
  }
}

// ---------------------------------------------------------------------------
// Scatter-max over edges via order-preserving float->u32 map + atomicMax
// (global_atomic_max_u32).  agg lives in cols [C,2C) of h [N,2C].
// ---------------------------------------------------------------------------
__device__ __forceinline__ unsigned f2ord(float f) {
  unsigned u = __float_as_uint(f);
  return (u & 0x80000000u) ? ~u : (u | 0x80000000u);
}
__device__ __forceinline__ float ord2f(unsigned u) {
  return __uint_as_float((u & 0x80000000u) ? (u & 0x7FFFFFFFu) : ~u);
}

template <int C>
__global__ void agg_init_kernel(unsigned* __restrict__ h, int N_) {
  int idx = blockIdx.x * blockDim.x + threadIdx.x;
  if (idx >= N_ * C) return;
  int n = idx / C, j = idx - n * C;
  h[(long)n * (2 * C) + C + j] = 0x007FFFFFu;   // ordered(-inf)
}

template <int C>
__global__ void scatter_max_kernel(const int* __restrict__ src,
                                   const int* __restrict__ dst,
                                   float* __restrict__ h, int E_) {
  int idx = blockIdx.x * blockDim.x + threadIdx.x;
  if (idx >= E_ * C) return;
  int e = idx / C, j = idx - e * C;
  int s = src[e], d = dst[e];
  unsigned key = f2ord(h[(long)s * (2 * C) + j]);
  atomicMax((unsigned*)h + (long)d * (2 * C) + C + j, key);
}

template <int C>
__global__ void agg_fix_kernel(float* __restrict__ h, int N_) {
  int idx = blockIdx.x * blockDim.x + threadIdx.x;
  if (idx >= N_ * C) return;
  int n = idx / C, j = idx - n * C;
  long o = (long)n * (2 * C) + C + j;
  float f = ord2f(((unsigned*)h)[o]);
  h[o] = (f > -3.0e38f && f < 3.0e38f) ? f : 0.0f;   // empty segment -> 0
}

// ---------------------------------------------------------------------------
// Per-polyline max-pool (32 contiguous nodes) + column sum-of-squares
// ---------------------------------------------------------------------------
__global__ void zero_kernel(float* p, int n) {
  int i = blockIdx.x * blockDim.x + threadIdx.x;
  if (i < n) p[i] = 0.0f;
}

__global__ void pool_kernel(const float* __restrict__ h, float* __restrict__ pooled,
                            float* __restrict__ sumsq) {
  int idx = blockIdx.x * blockDim.x + threadIdx.x;
  if (idx >= 4096 * 128) return;
  int p = idx >> 7, j = idx & 127;
  float m = -INFINITY;
#pragma unroll 4
  for (int i = 0; i < 32; ++i) m = fmaxf(m, h[((long)p * 32 + i) * 128 + j]);
  pooled[idx] = m;
  atomicAdd(&sumsq[j], m * m);
}

__global__ void colnorm_kernel(const float* __restrict__ pooled,
                               const float* __restrict__ sumsq,
                               float* __restrict__ feats) {
  int idx = blockIdx.x * blockDim.x + threadIdx.x;
  if (idx >= 4096 * 128) return;
  feats[idx] = pooled[idx] * rsqrtf(sumsq[idx & 127]);
}

// ---------------------------------------------------------------------------
// Q/K/V projections: feats[4096,128] @ W[128,64] + b, one wave per 16-row tile.
// A tile async-copied once; each W arrives via TDM per iteration.
// ---------------------------------------------------------------------------
__global__ void __launch_bounds__(32)
qkv_wmma_kernel(const float* __restrict__ feats,
                const float* __restrict__ Wq, const float* __restrict__ bq,
                const float* __restrict__ Wk, const float* __restrict__ bk,
                const float* __restrict__ Wv, const float* __restrict__ bv,
                float* __restrict__ Q, float* __restrict__ K, float* __restrict__ V) {
  __shared__ float sAf[16 * 128];
  __shared__ float sWf[128 * 64];
  const int lane = threadIdx.x & 31;
  const int half = lane >> 4, r = lane & 15;
  const int row0 = blockIdx.x * 16;

  {
    const char* gsrc = (const char*)(feats + (long)row0 * 128);
    const unsigned ldst = (unsigned)(uintptr_t)sAf;
    for (int c = lane; c < 512; c += 32)              // 16*128*4B / 16B
      async_ld_b128(ldst + c * 16, gsrc + c * 16);
  }
  wait_asynccnt0();

  const float* Ws[3] = {Wq, Wk, Wv};
  const float* bs[3] = {bq, bk, bv};
  float*       Os[3] = {Q, K, V};

  for (int m = 0; m < 3; ++m) {
    __syncthreads();
    tdm_load_2d(Ws[m], sWf, 64, 128, 64);
    __builtin_amdgcn_s_wait_tensorcnt(0);
    __syncthreads();
    v8f z = {};
    v8f acc[4] = {z, z, z, z};
#pragma unroll
    for (int kt = 0; kt < 4; ++kt) {
      v16h a = frag_a_f32(sAf, 128, kt * 32, lane, 128);
#pragma unroll
      for (int t = 0; t < 4; ++t)
        acc[t] = wmma_f16f32(a, frag_b_f32(sWf, 64, kt * 32, t * 16, lane,
                                           128, 64), acc[t]);
    }
#pragma unroll
    for (int t = 0; t < 4; ++t) {
      int col = t * 16 + r;
      float bb = bs[m][col];
#pragma unroll
      for (int i = 0; i < 8; ++i)
        Os[m][(long)(row0 + half * 8 + i) * 64 + col] = acc[t][i] + bb;
    }
  }
}

// ---------------------------------------------------------------------------
// Attention collapsed to the only consumed row (t=0):
//   score_s = q0 . k_s ; masked softmax over s ; ctx = sum_s attn_s * v_s
// ---------------------------------------------------------------------------
__global__ void attn_row0_kernel(const float* __restrict__ Q, const float* __restrict__ K,
                                 const float* __restrict__ V,
                                 const int* __restrict__ valid_len,
                                 float* __restrict__ ctx) {
  const int b = blockIdx.x;      // 16 batches
  const int s = threadIdx.x;     // 256 polylines
  __shared__ float attn[256];
  __shared__ float red[256];

  const float* q0 = Q + (long)b * 256 * 64;
  const float* ks = K + ((long)b * 256 + s) * 64;
  float dot = 0.f;
#pragma unroll 8
  for (int j = 0; j < 64; ++j) dot += q0[j] * ks[j];
  float score = (s <= valid_len[b]) ? dot : -1e-6f;   // reference's mask value

  red[s] = score;
  __syncthreads();
  for (int off = 128; off; off >>= 1) {
    if (s < off) red[s] = fmaxf(red[s], red[s + off]);
    __syncthreads();
  }
  float mx = red[0];
  __syncthreads();
  float ex = __expf(score - mx);
  attn[s] = ex;
  red[s] = ex;
  __syncthreads();
  for (int off = 128; off; off >>= 1) {
    if (s < off) red[s] += red[s + off];
    __syncthreads();
  }
  float inv = 1.0f / red[0];
  if (s < 64) {
    float acc = 0.f;
    for (int t = 0; t < 256; ++t)
      acc += attn[t] * V[((long)b * 256 + t) * 64 + s];
    ctx[b * 64 + s] = acc * inv;
  }
}

// ---------------------------------------------------------------------------
extern "C" void kernel_launch(void* const* d_in, const int* in_sizes, int n_in,
                              void* d_out, int out_size, void* d_ws, size_t ws_size,
                              hipStream_t stream) {
  (void)n_in; (void)out_size; (void)ws_size;

  const float* x        = (const float*)d_in[0];
  const int*   ei       = (const int*)d_in[1];
  const int*   validlen = (const int*)d_in[3];    // d_in[2] (cluster) implicit: node/32
  const int N = in_sizes[0] / 16;                 // 131072
  const int E = in_sizes[1] / 2;                  // 2097152
  const int* src = ei;
  const int* dst = ei + E;

  const float *l0W1=(const float*)d_in[4],  *l0b1=(const float*)d_in[5];
  const float *l0g =(const float*)d_in[6],  *l0be=(const float*)d_in[7];
  const float *l0W2=(const float*)d_in[8],  *l0b2=(const float*)d_in[9];
  const float *l1W1=(const float*)d_in[10], *l1b1=(const float*)d_in[11];
  const float *l1g =(const float*)d_in[12], *l1be=(const float*)d_in[13];
  const float *l1W2=(const float*)d_in[14], *l1b2=(const float*)d_in[15];
  const float *l2W1=(const float*)d_in[16], *l2b1=(const float*)d_in[17];
  const float *l2g =(const float*)d_in[18], *l2be=(const float*)d_in[19];
  const float *l2W2=(const float*)d_in[20], *l2b2=(const float*)d_in[21];
  const float *Wq=(const float*)d_in[22], *bq=(const float*)d_in[23];
  const float *Wk=(const float*)d_in[24], *bk=(const float*)d_in[25];
  const float *Wv=(const float*)d_in[26], *bv=(const float*)d_in[27];
  const float *pW1=(const float*)d_in[28], *pb1=(const float*)d_in[29];
  const float *pg =(const float*)d_in[30], *pbe=(const float*)d_in[31];
  const float *pW2=(const float*)d_in[32], *pb2=(const float*)d_in[33];

  // Workspace layout (floats): h1 [N,32] | h2 [N,64] | h3 [N,128];
  // small late-stage buffers alias h1 (dead after layer 1 consumes it).
  float* ws = (float*)d_ws;
  float* h1 = ws;
  float* h2 = h1 + (size_t)N * 32;
  float* h3 = h2 + (size_t)N * 64;
  float* pooled = h1;
  float* sumsq  = pooled + 4096 * 128;
  float* feats  = sumsq + 128;
  float* Qb     = feats + 4096 * 128;
  float* Kb     = Qb + 4096 * 64;
  float* Vb     = Kb + 4096 * 64;
  float* ctx    = Vb + 4096 * 64;

  const int TP = 256;
  dim3 w32(32);

  // ---- Layer 0 (16 -> 64 -> 16), concat with scatter-max -> h1 [N,32]
  mlp16_wmma_kernel<16,16,32><<<N/16, w32, 0, stream>>>(x, l0W1,l0b1,l0g,l0be,l0W2,l0b2, h1);
  agg_init_kernel<16><<<(N*16+TP-1)/TP, TP, 0, stream>>>((unsigned*)h1, N);
  scatter_max_kernel<16><<<(E*16+TP-1)/TP, TP, 0, stream>>>(src, dst, h1, E);
  agg_fix_kernel<16><<<(N*16+TP-1)/TP, TP, 0, stream>>>(h1, N);

  // ---- Layer 1 (32 -> 64 -> 32) -> h2 [N,64]
  mlp16_wmma_kernel<32,32,64><<<N/16, w32, 0, stream>>>(h1, l1W1,l1b1,l1g,l1be,l1W2,l1b2, h2);
  agg_init_kernel<32><<<(N*32+TP-1)/TP, TP, 0, stream>>>((unsigned*)h2, N);
  scatter_max_kernel<32><<<(E*32+TP-1)/TP, TP, 0, stream>>>(src, dst, h2, E);
  agg_fix_kernel<32><<<(N*32+TP-1)/TP, TP, 0, stream>>>(h2, N);

  // ---- Layer 2 (64 -> 64 -> 64) -> h3 [N,128]
  mlp16_wmma_kernel<64,64,128><<<N/16, w32, 0, stream>>>(h2, l2W1,l2b1,l2g,l2be,l2W2,l2b2, h3);
  agg_init_kernel<64><<<(N*64+TP-1)/TP, TP, 0, stream>>>((unsigned*)h3, N);
  scatter_max_kernel<64><<<(int)(((long)E*64+TP-1)/TP), TP, 0, stream>>>(src, dst, h3, E);
  agg_fix_kernel<64><<<(N*64+TP-1)/TP, TP, 0, stream>>>(h3, N);

  // ---- Max-pool per polyline + column-wise L2 normalization -> feats [4096,128]
  zero_kernel<<<1, 128, 0, stream>>>(sumsq, 128);
  pool_kernel<<<(4096*128)/TP, TP, 0, stream>>>(h3, pooled, sumsq);
  colnorm_kernel<<<(4096*128)/TP, TP, 0, stream>>>(pooled, sumsq, feats);

  // ---- Q/K/V projections (WMMA) -> [4096,64] each
  qkv_wmma_kernel<<<4096/16, w32, 0, stream>>>(feats, Wq,bq, Wk,bk, Wv,bv, Qb,Kb,Vb);

  // ---- Attention, only t=0 row consumed -> ctx [16,64]
  attn_row0_kernel<<<16, 256, 0, stream>>>(Qb, Kb, Vb, validlen, ctx);

  // ---- Head: relu(LN(ctx@pW1+pb1))@pW2+pb2 -> out [16,60]
  mlp16_wmma_kernel<64,60,60><<<1, w32, 0, stream>>>(ctx, pW1,pb1,pg,pbe,pW2,pb2,
                                                     (float*)d_out);
}
